// AdaModConv1D_90099823935851
// MI455X (gfx1250) — matrix operand order repaired and checked by reading, not compiled, not run on hardware
//
#include <hip/hip_runtime.h>

typedef __attribute__((ext_vector_type(16))) __bf16        v16bf;
typedef __attribute__((ext_vector_type(8)))  float         v8f;
typedef __attribute__((ext_vector_type(4)))  unsigned int  u32x4;
typedef __attribute__((ext_vector_type(4)))  float         f32x4;
typedef __attribute__((ext_vector_type(4)))  __bf16        bf16x4;

#define B_      8
#define L_      65536
#define C_      64
#define F_      64
#define KW      3
#define DL_     256
#define KTOT    (KW * C_)      // 192  (reduction dim of implicit GEMM)
#define TILE_M  256            // output rows per block (8 waves x 32 rows)
#define XROWS   (TILE_M + 2)   // halo of 1 row each side (kernel size 3, SAME)
#define XPITCH  72             // padded row pitch (bf16 elems) for LDS bank spread
#define WPITCH  200            // padded K pitch (bf16 elems) for transposed weights

union FragBF { v16bf v; u32x4 q[2]; };

// ---------------------------------------------------------------------------
// Kernel 1: per-sample modulation scale s[b,f] and demodulation d[b,f].
//   s = softplus(ltnt @ Wd + bd) + 1
//   d = rsqrt( sum_{K} (kernel[K,f] * s[b, K%64])^2 + eps )
// ws layout: ws[0..511] = s, ws[512..1023] = d   (float)
// ---------------------------------------------------------------------------
__global__ __launch_bounds__(256) void adamod_prep(
    const float* __restrict__ ltnt, const float* __restrict__ kern,
    const float* __restrict__ Wd,   const float* __restrict__ bd,
    float* __restrict__ ws) {
  __shared__ float sAll[B_ * F_];
  const int tid = threadIdx.x;

  for (int p = tid; p < B_ * F_; p += 256) {
    const int b = p >> 6, f = p & 63;
    float acc = bd[f];
    for (int i = 0; i < DL_; ++i)
      acc = fmaf(ltnt[b * DL_ + i], Wd[i * F_ + f], acc);
    const float sp = (acc > 20.f) ? acc : log1pf(expf(acc));
    const float s  = sp + 1.f;
    sAll[p] = s;
    ws[p]   = s;
  }
  __syncthreads();

  for (int p = tid; p < B_ * F_; p += 256) {
    const int b = p >> 6, f = p & 63;
    float sum = 0.f;
    for (int K = 0; K < KTOT; ++K) {
      const float w  = kern[K * F_ + f];
      const float sc = sAll[b * C_ + (K & 63)];
      sum = fmaf(w * sc, w * sc, sum);
    }
    ws[B_ * F_ + p] = rsqrtf(sum + 1e-8f);
  }
}

// ---------------------------------------------------------------------------
// Kernel 2: implicit-GEMM conv via bf16 WMMA with hi/lo split precision.
//   y[b,l,f] = d[b,f] * sum_{k,c} (x[b,l+k-1,c]*s[b,c]) * w[k,c,f]
// Block: one batch b, 256 rows. Wave: 32 rows x 64 f (two 16x64 sub-tiles,
// sharing B fragments -> 1:1 ds_load_b128 : v_wmma ratio).
// ---------------------------------------------------------------------------
__global__ __launch_bounds__(256) void adamod_conv(
    const float* __restrict__ x, const float* __restrict__ kern,
    const float* __restrict__ ws, float* __restrict__ out) {
  __shared__ __bf16 xh[XROWS * XPITCH];
  __shared__ __bf16 xl[XROWS * XPITCH];
  __shared__ __bf16 wh[F_ * WPITCH];
  __shared__ __bf16 wl[F_ * WPITCH];
  __shared__ float  ss[C_];

  const int tid = threadIdx.x;
  const int b   = blockIdx.x / (L_ / TILE_M);
  const int m0  = (blockIdx.x % (L_ / TILE_M)) * TILE_M;

  if (tid < C_) ss[tid] = ws[b * C_ + tid];
  __syncthreads();

  // Stage weights transposed [f][K], split hi/lo. Global reads are b128
  // (4 consecutive f); LDS stores are strided b16.
  // kernel flat index = (k*64 + c)*64 + f = K*64 + f.
  for (int idx = tid; idx < (KTOT * F_) / 4; idx += 256) {
    const int K  = idx >> 4;
    const int f4 = (idx & 15) * 4;
    const f32x4 v = *(const f32x4*)(&kern[K * F_ + f4]);
#pragma unroll
    for (int j = 0; j < 4; ++j) {
      const __bf16 hi = (__bf16)v[j];
      const __bf16 lo = (__bf16)(v[j] - (float)hi);
      wh[(f4 + j) * WPITCH + K] = hi;
      wl[(f4 + j) * WPITCH + K] = lo;
    }
  }
  // Stage modulated input tile (rows m0-1 .. m0+TILE_M), split hi/lo.
  // b128 global reads, b64 LDS stores.
  const long xbase = (long)b * L_ * C_;
  for (int idx = tid; idx < XROWS * (C_ / 4); idx += 256) {
    const int r  = idx >> 4;
    const int c4 = (idx & 15) * 4;
    const int g  = m0 - 1 + r;
    f32x4 v = {0.f, 0.f, 0.f, 0.f};
    if (g >= 0 && g < L_) v = *(const f32x4*)(&x[xbase + (long)g * C_ + c4]);
    bf16x4 hi, lo;
#pragma unroll
    for (int j = 0; j < 4; ++j) {
      const float  vv = v[j] * ss[c4 + j];
      const __bf16 h  = (__bf16)vv;
      hi[j] = h;
      lo[j] = (__bf16)(vv - (float)h);
    }
    *(bf16x4*)(&xh[r * XPITCH + c4]) = hi;
    *(bf16x4*)(&xl[r * XPITCH + c4]) = lo;
  }
  __syncthreads();

  const int wave = tid >> 5;
  const int lane = tid & 31;
  const int half = lane >> 4;   // lane half: selects K sub-range per ISA layout
  const int lr   = lane & 15;

  v8f acc[2][4] = {{v8f{}, v8f{}, v8f{}, v8f{}}, {v8f{}, v8f{}, v8f{}, v8f{}}};

  // K loop: 192 = 6 chunks of 32. Chunk q: k = q>>1 (constant), c in [(q&1)*32,+32)
#pragma unroll
  for (int q = 0; q < 6; ++q) {
    const int k  = q >> 1;
    const int cb = (q & 1) * 32;

    // Two stacked A sub-tiles (rows wave*32+0..15 and +16..31).
    FragBF a0h, a0l, a1h, a1l;
    {
      const int r0 = wave * 32 + lr + k;           // sub-tile 0 LDS row
      const int o0 = r0 * XPITCH + cb + 8 * half;  // elems 0..7 / 8..15 at +16
      a0h.q[0] = *(const u32x4*)(&xh[o0]);
      a0h.q[1] = *(const u32x4*)(&xh[o0 + 16]);
      a0l.q[0] = *(const u32x4*)(&xl[o0]);
      a0l.q[1] = *(const u32x4*)(&xl[o0 + 16]);
      const int o1 = o0 + 16 * XPITCH;             // sub-tile 1: +16 rows
      a1h.q[0] = *(const u32x4*)(&xh[o1]);
      a1h.q[1] = *(const u32x4*)(&xh[o1 + 16]);
      a1l.q[0] = *(const u32x4*)(&xl[o1]);
      a1l.q[1] = *(const u32x4*)(&xl[o1 + 16]);
    }

#pragma unroll
    for (int t = 0; t < 4; ++t) {
      const int f    = t * 16 + lr;
      const int koff = f * WPITCH + q * 32 + 8 * half;  // K elems 0..7 / 8..15 at +16
      FragBF bh, bl;
      bh.q[0] = *(const u32x4*)(&wh[koff]);
      bh.q[1] = *(const u32x4*)(&wh[koff + 16]);
      bl.q[0] = *(const u32x4*)(&wl[koff]);
      bl.q[1] = *(const u32x4*)(&wl[koff + 16]);

      // split precision: x*w ~= xh*wh + xh*wl + xl*wh  (f32 accumulate)
      acc[0][t] = __builtin_amdgcn_wmma_f32_16x16x32_bf16(
          false, a0h.v, false, bh.v, (short)0, acc[0][t], false, false);
      acc[0][t] = __builtin_amdgcn_wmma_f32_16x16x32_bf16(
          false, a0h.v, false, bl.v, (short)0, acc[0][t], false, false);
      acc[0][t] = __builtin_amdgcn_wmma_f32_16x16x32_bf16(
          false, a0l.v, false, bh.v, (short)0, acc[0][t], false, false);

      acc[1][t] = __builtin_amdgcn_wmma_f32_16x16x32_bf16(
          false, a1h.v, false, bh.v, (short)0, acc[1][t], false, false);
      acc[1][t] = __builtin_amdgcn_wmma_f32_16x16x32_bf16(
          false, a1h.v, false, bl.v, (short)0, acc[1][t], false, false);
      acc[1][t] = __builtin_amdgcn_wmma_f32_16x16x32_bf16(
          false, a1l.v, false, bh.v, (short)0, acc[1][t], false, false);
    }
  }

  // Epilogue: scale by d[b,f], store fp32.
  // C/D layout: lane (half,lr) -> N = lr; VGPR r -> M = r + 8*half.
  const long obase = (long)b * L_ * F_;
#pragma unroll
  for (int t = 0; t < 4; ++t) {
    const int   f      = t * 16 + lr;
    const float dscale = ws[B_ * F_ + b * F_ + f];
#pragma unroll
    for (int s = 0; s < 2; ++s) {
#pragma unroll
      for (int r = 0; r < 8; ++r) {
        const int m = m0 + wave * 32 + s * 16 + half * 8 + r;
        out[obase + (long)m * F_ + f] = acc[s][t][r] * dscale;
      }
    }
  }
}

// ---------------------------------------------------------------------------
extern "C" void kernel_launch(void* const* d_in, const int* in_sizes, int n_in,
                              void* d_out, int out_size, void* d_ws, size_t ws_size,
                              hipStream_t stream) {
  const float* data = (const float*)d_in[0];  // [B, L, C]
  const float* ltnt = (const float*)d_in[1];  // [B, DL]
  const float* kern = (const float*)d_in[2];  // [K, C, F]
  const float* Wd   = (const float*)d_in[3];  // [DL, F]
  const float* bd   = (const float*)d_in[4];  // [F]
  float*       out  = (float*)d_out;          // [B, L, F]
  float*       ws   = (float*)d_ws;           // 1024 floats: s then d

  adamod_prep<<<1, 256, 0, stream>>>(ltnt, kern, Wd, bd, ws);
  adamod_conv<<<B_ * (L_ / TILE_M), 256, 0, stream>>>(data, kern, ws, out);
}
